// Lstm_18665927868754
// MI455X (gfx1250) — compile-verified, matrix operand order checked by reading, not run on hardware
//
#include <hip/hip_runtime.h>
#include <stdint.h>

#define T_STEPS 2048
#define BATCH   4096
#define HID     20
#define NTILES  8      // padded gate columns 4*32 = 128 -> 8 N-tiles of 16
#define WAVES   8      // waves per block (block = 256 threads, 128 batch rows)
#define CHUNK   32     // timesteps of u prefetched per async chunk

typedef __attribute__((ext_vector_type(16))) __bf16 v16bf;
typedef __attribute__((ext_vector_type(8)))  float  v8f;

union V16 { uint32_t u[8]; v16bf v; };

__device__ __forceinline__ uint32_t f2bf(float x) {
  uint32_t b = __float_as_uint(x);
  return (b + 0x7FFFu + ((b >> 16) & 1u)) >> 16;   // round-to-nearest-even bf16
}

#if __has_builtin(__builtin_amdgcn_tanhf)
// gfx1250 V_TANH_F32 (TRANS pipe, co-executes with VALU)
__device__ __forceinline__ float fast_tanh(float x) {
  return __builtin_amdgcn_tanhf(x);
}
__device__ __forceinline__ float fast_sigmoid(float x) {
  return fmaf(__builtin_amdgcn_tanhf(x * 0.5f), 0.5f, 0.5f);
}
#else
__device__ __forceinline__ float fast_tanh(float x) {
  float e = __builtin_amdgcn_exp2f(x * 2.88539008f);    // exp(2x)
  return (e - 1.0f) * __builtin_amdgcn_rcpf(e + 1.0f);
}
__device__ __forceinline__ float fast_sigmoid(float x) {
  float e = __builtin_amdgcn_exp2f(x * -1.44269504f);   // exp(-x)
  return __builtin_amdgcn_rcpf(1.0f + e);
}
#endif

__device__ __forceinline__ void async_b128(uint32_t lds_off, const void* g) {
  asm volatile("global_load_async_to_lds_b128 %0, %1, off"
               :: "v"(lds_off), "v"(g) : "memory");
}

__global__ __launch_bounds__(WAVES * 32, 1) void lstm_wmma_kernel(
    const float* __restrict__ u, const float* __restrict__ Wih,
    const float* __restrict__ Whh, const float* __restrict__ Wout,
    float* __restrict__ out)
{
  // per-wave u stream (double buffered) + per-wave h transpose scratch
  __shared__ __align__(16) float    sh_u[2][WAVES][CHUNK][16];   // 32 KB
  __shared__ __align__(16) uint16_t sh_h[WAVES][16][32];         // 8 KB

  const int lane = threadIdx.x & 31;
  const int wave = threadIdx.x >> 5;
  const int half = lane >> 4;       // which 16-lane half
  const int lcol = lane & 15;
  const int b0   = blockIdx.x * (WAVES * 16) + wave * 16;

  // lanes 0-15 hold A's K=20,21 slot in VGPR6 -> inject u there (branchless)
  const uint32_t injmask = (half == 0) ? 0xFFFFFFFFu : 0u;

  // ---- combined recurrent+input weights as 8 padded bf16 B-tiles ----
  // B[k][n]: k<20 -> W_hh[gate_row][k]; k==20 -> W_ih[gate_row] (input
  // projection folded into the same WMMA via A's K=20 slot = u_t); else 0.
  // Assumed 16-bit B layout (mirror of documented A layout, M->N):
  //   lanes 0-15: N=lcol, VGPR v<4 holds K=2v,2v+1; v>=4 holds K=2v+8,2v+9
  //   lanes 16-31: same N, K offset +8
  V16 Bt[NTILES];
#pragma unroll
  for (int t = 0; t < NTILES; ++t) {
    const int n  = t * 16 + lcol;   // padded gate column
    const int G  = n >> 5;          // gate: 0=i 1=f 2=g 3=o
    const int wi = n & 31;          // index within gate (valid < 20)
    const float wihv = (wi < HID) ? Wih[G * HID + wi] : 0.0f;
#pragma unroll
    for (int v = 0; v < 8; ++v) {
      uint32_t pk = 0;
#pragma unroll
      for (int s = 0; s < 2; ++s) {
        const int k = 2 * v + s + (v >= 4 ? 8 : 0) + 8 * half;
        float val;
        if (k < HID)       val = (wi < HID) ? Whh[(G * HID + wi) * HID + k] : 0.0f;
        else if (k == HID) val = wihv;      // input-projection row
        else               val = 0.0f;
        pk |= f2bf(val) << (16 * s);
      }
      Bt[t].u[v] = pk;
    }
  }

  float wo[2];
#pragma unroll
  for (int hh = 0; hh < 2; ++hh) {
    const int n = hh * 16 + lcol;
    wo[hh] = (n < HID) ? Wout[n] : 0.0f;
  }

  // zero h scratch (h0 = 0) and cell state
  {
    uint4 z = make_uint4(0, 0, 0, 0);
    uint4* p = (uint4*)&sh_h[wave][0][0];
    p[lane] = z; p[lane + 32] = z;
  }
  float cst[2][8];
#pragma unroll
  for (int hh = 0; hh < 2; ++hh)
#pragma unroll
    for (int r = 0; r < 8; ++r) cst[hh][r] = 0.0f;

  // async u prefetch: each chunk = 32 rows x 64B, 4 x b128 per lane-set
  const int prow = lane >> 2;        // 0..7
  const int pcol = (lane & 3) * 4;   // float column 0,4,8,12

  auto prefetch = [&](int chunk, int pb) {
    const float* g = u + (size_t)(chunk * CHUNK + prow) * BATCH + b0 + pcol;
    uint32_t l = (uint32_t)(uintptr_t)&sh_u[pb][wave][prow][pcol];
#pragma unroll
    for (int j = 0; j < 4; ++j) {
      async_b128(l, (const void*)g);
      g += (size_t)8 * BATCH;
      l += 8 * 64;
    }
  };

  prefetch(0, 0);

  const v8f zc = {};   // inline-0 WMMA accumulator

  const int NCHUNK = T_STEPS / CHUNK;
  for (int c = 0; c < NCHUNK; ++c) {
    if (c + 1 < NCHUNK) {
      prefetch(c + 1, (c + 1) & 1);
      asm volatile("s_wait_asynccnt 4" ::: "memory");  // oldest 4 (this chunk) done
    } else {
      asm volatile("s_wait_asynccnt 0" ::: "memory");
    }
    const int rb = c & 1;

    for (int tt = 0; tt < CHUNK; ++tt) {
      const int t = c * CHUNK + tt;

      // A := [ h_prev | u_t ] (bf16, row-major in LDS) in WMMA A layout.
      // Lane L<16 holds row M=L; its VGPR6 = K 20,21 -> inject u_t[b0+L] at K=20.
      V16 A;
      {
        const char* base = (const char*)&sh_h[wave][0][0] + lcol * 64 + half * 16;
        uint4 a0 = *(const uint4*)(base);        // K 0..7   (or 8..15 upper half)
        uint4 a1 = *(const uint4*)(base + 32);   // K 16..23 (or 24..31)
        float uval = sh_u[rb][wave][tt][lcol];   // u_t for batch row lcol
        uint32_t inj = f2bf(uval);               // K=20 bf16, K=21 stays 0
        A.u[0] = a0.x; A.u[1] = a0.y; A.u[2] = a0.z; A.u[3] = a0.w;
        A.u[4] = a1.x; A.u[5] = a1.y;
        A.u[6] = (inj & injmask) | (a1.z & ~injmask);   // branchless (v_bfi)
        A.u[7] = a1.w;
      }

      float part[8];
#pragma unroll
      for (int r = 0; r < 8; ++r) part[r] = 0.0f;

#pragma unroll
      for (int hh = 0; hh < 2; ++hh) {
        v8f gI = __builtin_amdgcn_wmma_f32_16x16x32_bf16(false, A.v, false, Bt[0 + hh].v, (short)0, zc, false, false);
        v8f gF = __builtin_amdgcn_wmma_f32_16x16x32_bf16(false, A.v, false, Bt[2 + hh].v, (short)0, zc, false, false);
        v8f gG = __builtin_amdgcn_wmma_f32_16x16x32_bf16(false, A.v, false, Bt[4 + hh].v, (short)0, zc, false, false);
        v8f gO = __builtin_amdgcn_wmma_f32_16x16x32_bf16(false, A.v, false, Bt[6 + hh].v, (short)0, zc, false, false);

#pragma unroll
        for (int r = 0; r < 8; ++r) {
          float ig = fast_sigmoid(gI[r]);
          float fg = fast_sigmoid(gF[r]);
          float gg = fast_tanh(gG[r]);
          float og = fast_sigmoid(gO[r]);
          float cn = fg * cst[hh][r] + ig * gg;
          cst[hh][r] = cn;
          float hn = og * fast_tanh(cn);
          part[r] = fmaf(hn, wo[hh], part[r]);
          // write h back (row-major bf16) for next step's A
          sh_h[wave][r + 8 * half][lcol + 16 * hh] = (uint16_t)f2bf(hn);
        }
      }

      // y[t, b0+M] = sum_N h*W_out : butterfly reduce across each 16-lane half
#pragma unroll
      for (int r = 0; r < 8; ++r) {
#pragma unroll
        for (int m = 1; m <= 8; m <<= 1)
          part[r] += __shfl_xor(part[r], m, 32);
      }
      const int key = lcol & 7;
      float q0 = (key & 1) ? part[1] : part[0];
      float q1 = (key & 1) ? part[3] : part[2];
      float q2 = (key & 1) ? part[5] : part[4];
      float q3 = (key & 1) ? part[7] : part[6];
      float r0 = (key & 2) ? q1 : q0;
      float r1 = (key & 2) ? q3 : q2;
      float sel = (key & 4) ? r1 : r0;               // lane holds y for its (half, key)
      float yv = __shfl(sel, lane + (lane & 8), 32); // pull upper-half rows 8..15 down
      if (lane < 16) out[(size_t)t * BATCH + b0 + lane] = yv;
    }
  }
}

extern "C" void kernel_launch(void* const* d_in, const int* in_sizes, int n_in,
                              void* d_out, int out_size, void* d_ws, size_t ws_size,
                              hipStream_t stream) {
  const float* u    = (const float*)d_in[0];
  const float* Wih  = (const float*)d_in[1];
  const float* Whh  = (const float*)d_in[2];
  const float* Wout = (const float*)d_in[3];
  float* out = (float*)d_out;

  dim3 grid(BATCH / (WAVES * 16));   // 32 blocks x 8 waves = 256 batch tiles
  dim3 block(WAVES * 32);
  lstm_wmma_kernel<<<grid, block, 0, stream>>>(u, Wih, Whh, Wout, out);
}
